// WindowAttention_13632226198199
// MI455X (gfx1250) — compile-verified, matrix operand order checked by reading, not run on hardware
//
#include <hip/hip_runtime.h>
#include <hip/hip_bf16.h>
#include <math.h>

typedef __attribute__((ext_vector_type(16))) _Float16 v16h;
typedef __attribute__((ext_vector_type(8)))  _Float16 v8h;
typedef __attribute__((ext_vector_type(4)))  _Float16 v4h;
typedef __attribute__((ext_vector_type(8)))  float    v8f;

#define B_TOT   4096
#define N_TOK   64
#define C_DIM   128
#define NHEAD   4
#define HD      32
#define NW      2048
#define QSCALE  0.1767766952966369f   /* 32^-0.5 */

// LDS row strides (halves), padded +8 to spread banks; all stay 16B aligned.
#define SX  136   /* 64 x 128 buffer (y -> q -> o) */
#define SK  136   /* 64 x 128 k (head-major cols)  */
#define SVT 72    /* 128 x 64 v transposed         */
#define SP  72    /* per-wave 16 x 64 prob strips  */

// ---------------------------------------------------------------------------
// WMMA helpers
// ---------------------------------------------------------------------------
__device__ __forceinline__ v8f wmma_f16(v16h a, v16h b, v8f c) {
  return __builtin_amdgcn_wmma_f32_16x16x32_f16(
      /*neg_a=*/false, a, /*neg_b=*/false, b,
      /*c_mod=*/(short)0, c, /*reuse_a=*/false, /*reuse_b=*/false);
}

// Load one 16x32 f16 operand tile (A or B) from a row-major buffer.
// Per-lane: row = caller-supplied; K layout per CDNA5 16-bit operand spec:
//   lanes 0-15  hold K = {0..7, 16..23}, lanes 16-31 hold K = {8..15, 24..31}
// -> two contiguous 16-byte loads at colbase+kb and colbase+kb+16.
__device__ __forceinline__ v16h load_tile16(const _Float16* base, int row,
                                            int stride, int colbase, int kb) {
  const _Float16* p = base + row * stride + colbase + kb;
  v8h lo = *(const v8h*)(p);
  v8h hi = *(const v8h*)(p + 16);
  v16h r;
#pragma unroll
  for (int i = 0; i < 8; ++i) { r[i] = lo[i]; r[i + 8] = hi[i]; }
  return r;
}

// ---------------------------------------------------------------------------
// Kernel 0: convert weights f32 -> f16 into workspace
//   ws[0       .. 32767]  = qkv_w  (256 x 128)
//   ws[32768   .. 49151]  = proj_w (128 x 128)
// ---------------------------------------------------------------------------
__global__ __launch_bounds__(256)
void wconvert_kernel(const float* __restrict__ qkv_w,
                     const float* __restrict__ proj_w,
                     _Float16* __restrict__ ws) {
  int i = blockIdx.x * 256 + threadIdx.x;
  if (i < 32768)      ws[i] = (_Float16)qkv_w[i];
  else if (i < 49152) ws[i] = (_Float16)proj_w[i - 32768];
}

// ---------------------------------------------------------------------------
// Main kernel: one block per window b, 256 threads = 8 waves (wave32)
// ---------------------------------------------------------------------------
__global__ __launch_bounds__(256)
void win_attn_kernel(const float* __restrict__ x,
                     const float* __restrict__ y,
                     const float* __restrict__ Pth,
                     const float* __restrict__ mask,
                     const float* __restrict__ qkv_b,
                     const float* __restrict__ proj_b,
                     const _Float16* __restrict__ wQkv,   // 256x128 f16
                     const _Float16* __restrict__ wProj,  // 128x128 f16
                     const float* __restrict__ lambda_p,
                     float* __restrict__ out) {
  extern __shared__ _Float16 smem[];
  _Float16* sK  = smem;                     // 64*SK
  _Float16* sVT = sK  + 64 * SK;            // 128*SVT
  _Float16* sX  = sVT + 128 * SVT;          // 64*SX  (y -> q -> o)
  _Float16* sPr = sX  + 64 * SX;            // 8 waves * 16*SP

  const int b    = blockIdx.x;
  const int tid  = threadIdx.x;
  const int wave = tid >> 5;
  const int lane = tid & 31;
  const int ln   = lane & 15;
  const int hb   = (lane & 16) ? 8 : 0;     // row-half offset == k-base offset
  const float lam = lambda_p[0];

  const float* thermal0 = Pth + (size_t)b * NHEAD * 4096;
  // Prefetch this block's thermal bias (16384 floats = 64KB) into cache.
  __builtin_prefetch(thermal0 + tid * 64, 0, 1);

  // ---------------- Phase 1: y -> sX (f16) --------------------------------
  const float* yb = y + (size_t)b * 8192;
#pragma unroll
  for (int it = 0; it < 8; ++it) {
    int off = it * 1024 + tid * 4;                 // linear (n, c)
    float4 f = *(const float4*)(yb + off);
    int n = off >> 7, c = off & 127;
    v4h h4 = { (_Float16)f.x, (_Float16)f.y, (_Float16)f.z, (_Float16)f.w };
    *(v4h*)(sX + n * SX + c) = h4;
  }
  __syncthreads();

  // ---------------- Phase 2: KV = Y @ Wqkv^T + b --------------------------
  // 64x256 output: waves 0-3 -> K half (cols 0..127, already head-major),
  //                waves 4-7 -> V half (cols 128..255), stored transposed.
  {
    const int ms = wave & 3;
    const int ng = (wave >> 2) * 8;            // n-tile group base (8 tiles)
    v8f acc[8] = {};
#pragma unroll
    for (int kk = 0; kk < 4; ++kk) {
      v16h a = load_tile16(sX, ms * 16 + ln, SX, kk * 32, hb);
#pragma unroll
      for (int nt = 0; nt < 8; ++nt) {
        v16h bm = load_tile16(wQkv, (ng + nt) * 16 + ln, 128, kk * 32, hb);
        acc[nt] = wmma_f16(a, bm, acc[nt]);
      }
    }
#pragma unroll
    for (int nt = 0; nt < 8; ++nt) {
      const int cout = (ng + nt) * 16 + ln;
      const float bias = qkv_b[cout];
#pragma unroll
      for (int r = 0; r < 8; ++r) {
        int nrow = ms * 16 + r + hb;
        float vv = acc[nt][r] + bias;
        if (cout < 128) sK[nrow * SK + cout] = (_Float16)vv;           // K
        else            sVT[(cout - 128) * SVT + nrow] = (_Float16)vv; // V^T
      }
    }
  }
  __syncthreads();

  // ---------------- Phase 3: x -> sX as Q (axis-mixing reshape + scale) ---
  const float* xb = x + (size_t)b * 8192;
#pragma unroll
  for (int it = 0; it < 8; ++it) {
    int s = it * 1024 + tid * 4;               // flat source index
    float4 f = *(const float4*)(xb + s);
    int h = s >> 11, np = (s >> 5) & 63, d = s & 31;
    int dest = np * SX + h * 32 + d;           // q[np][h*32+d]
    v4h h4 = { (_Float16)(f.x * QSCALE), (_Float16)(f.y * QSCALE),
               (_Float16)(f.z * QSCALE), (_Float16)(f.w * QSCALE) };
    *(v4h*)(sX + dest) = h4;
  }
  __syncthreads();

  // ---------------- Phase 4+5: scores, softmax, O = P @ V -----------------
  const float* mbase = mask + (size_t)(b & (NW - 1)) * 4096;
  v8f oacc[2][2] = {};
  _Float16* slot = sPr + wave * 16 * SP;

#pragma unroll
  for (int rnd = 0; rnd < 2; ++rnd) {
    const int strip = wave + rnd * 8;          // 16 strips: (head, row-strip)
    const int h  = strip >> 2;
    const int ms = strip & 3;
    const float* tbase = thermal0 + (size_t)h * 4096;

    // S strip = Q[ms] @ K_h^T : 4 tiles, single WMMA each (K = hd = 32)
    v8f sacc[4];
    {
      v16h a = load_tile16(sX, ms * 16 + ln, SX, h * 32, hb);
      v8f z = {};
#pragma unroll
      for (int nt = 0; nt < 4; ++nt) {
        v16h bm = load_tile16(sK, nt * 16 + ln, SK, h * 32, hb);
        sacc[nt] = wmma_f16(a, bm, z);
      }
    }
    // additive mask + lambda * thermal, straight from global
#pragma unroll
    for (int nt = 0; nt < 4; ++nt) {
      int j = nt * 16 + ln;
#pragma unroll
      for (int r = 0; r < 8; ++r) {
        int i = ms * 16 + r + hb;
        sacc[nt][r] += mbase[i * 64 + j] + lam * tbase[i * 64 + j];
      }
    }
    // register softmax: rows live in 16-lane halves of one VGPR across 4 tiles
#pragma unroll
    for (int r = 0; r < 8; ++r) {
      float mx = sacc[0][r];
#pragma unroll
      for (int nt = 1; nt < 4; ++nt) mx = fmaxf(mx, sacc[nt][r]);
#pragma unroll
      for (int m = 1; m < 16; m <<= 1) mx = fmaxf(mx, __shfl_xor(mx, m, 32));
      float e[4], sum = 0.f;
#pragma unroll
      for (int nt = 0; nt < 4; ++nt) { e[nt] = __expf(sacc[nt][r] - mx); sum += e[nt]; }
#pragma unroll
      for (int m = 1; m < 16; m <<= 1) sum += __shfl_xor(sum, m, 32);
      float inv = 1.0f / sum;
      int lm = r + hb;
#pragma unroll
      for (int nt = 0; nt < 4; ++nt)
        slot[lm * SP + nt * 16 + ln] = (_Float16)(e[nt] * inv);
    }
    __syncthreads();   // fence: P staged in LDS before operand reload

    // O strip += P(16x64) @ V_h(64x32): K = 64 -> 2 chunks, 2 d-tiles
#pragma unroll
    for (int kk = 0; kk < 2; ++kk) {
      v16h a = load_tile16(slot, ln, SP, kk * 32, hb);
#pragma unroll
      for (int dt = 0; dt < 2; ++dt) {
        v16h bm = load_tile16(sVT, h * 32 + dt * 16 + ln, SVT, kk * 32, hb);
        oacc[rnd][dt] = wmma_f16(a, bm, oacc[rnd][dt]);
      }
    }
    __syncthreads();   // slot reused next round
  }

  // ---------------- Store O (head concat) into sX -------------------------
  __syncthreads();     // all Q reads done; safe to overwrite sX
#pragma unroll
  for (int rnd = 0; rnd < 2; ++rnd) {
    const int strip = wave + rnd * 8;
    const int h  = strip >> 2;
    const int ms = strip & 3;
#pragma unroll
    for (int dt = 0; dt < 2; ++dt)
#pragma unroll
      for (int r = 0; r < 8; ++r) {
        int i = ms * 16 + r + hb;
        sX[i * SX + h * 32 + dt * 16 + ln] = (_Float16)oacc[rnd][dt][r];
      }
  }
  __syncthreads();

  // ---------------- Phase 6: out = O @ Wproj^T + b ------------------------
  {
    const int ms = wave & 3;
    const int ng = (wave >> 2) * 4;            // 8 n-tiles split 4+4
    float* ob = out + (size_t)b * 8192;
    v8f acc[4] = {};
#pragma unroll
    for (int kk = 0; kk < 4; ++kk) {
      v16h a = load_tile16(sX, ms * 16 + ln, SX, kk * 32, hb);
#pragma unroll
      for (int nt = 0; nt < 4; ++nt) {
        v16h bm = load_tile16(wProj, (ng + nt) * 16 + ln, 128, kk * 32, hb);
        acc[nt] = wmma_f16(a, bm, acc[nt]);
      }
    }
#pragma unroll
    for (int nt = 0; nt < 4; ++nt) {
      const int c = (ng + nt) * 16 + ln;
      const float bias = proj_b[c];
#pragma unroll
      for (int r = 0; r < 8; ++r) {
        int i = ms * 16 + r + hb;
        ob[i * 128 + c] = acc[nt][r] + bias;
      }
    }
  }
}

// ---------------------------------------------------------------------------
extern "C" void kernel_launch(void* const* d_in, const int* in_sizes, int n_in,
                              void* d_out, int out_size, void* d_ws, size_t ws_size,
                              hipStream_t stream) {
  const float* x      = (const float*)d_in[0];
  const float* y      = (const float*)d_in[1];
  const float* Pth    = (const float*)d_in[2];
  const float* mask   = (const float*)d_in[3];
  const float* qkv_w  = (const float*)d_in[4];
  const float* qkv_b  = (const float*)d_in[5];
  const float* proj_w = (const float*)d_in[6];
  const float* proj_b = (const float*)d_in[7];
  const float* lam    = (const float*)d_in[8];

  _Float16* wsH = (_Float16*)d_ws;   // 49152 halves = 96KB scratch

  wconvert_kernel<<<192, 256, 0, stream>>>(qkv_w, proj_w, wsH);

  const size_t lds_bytes =
      (size_t)(64 * SK + 128 * SVT + 64 * SX + 8 * 16 * SP) * sizeof(_Float16);
  win_attn_kernel<<<B_TOT, 256, lds_bytes, stream>>>(
      x, y, Pth, mask, qkv_b, proj_b, wsH, wsH + 32768, lam, (float*)d_out);
}
